// B_51977694216330
// MI455X (gfx1250) — matrix operand order repaired and checked by reading, not compiled, not run on hardware
//
#include <hip/hip_runtime.h>
#include <hip/hip_bf16.h>
#include <math.h>

typedef __attribute__((ext_vector_type(16))) _Float16 v16h;
typedef __attribute__((ext_vector_type(8)))  float    v8f;
typedef unsigned int uvec4_t __attribute__((ext_vector_type(4)));
typedef unsigned int uvec2_t __attribute__((ext_vector_type(2)));

#define B_   64
#define L_   256
#define H_   768
#define S_   320
#define H2_  1536
#define OA_  3840   // 5*H

// ---------------------------------------------------------------------------
// WMMA helpers (CDNA5 wave32, V_WMMA_F32_16X16X32_F16)
// ---------------------------------------------------------------------------
__device__ __forceinline__ v8f vzero8() {
    v8f z = {0.f,0.f,0.f,0.f,0.f,0.f,0.f,0.f};
    return z;
}

__device__ __forceinline__ v8f wmma_f16(v16h a, v16h b, v8f c) {
    return __builtin_amdgcn_wmma_f32_16x16x32_f16(
        /*neg_a=*/false, a, /*neg_b=*/false, b,
        /*c_mod=*/(short)0, c, /*reuse_a=*/false, /*reuse_b=*/false);
}

// Async global->LDS copy, 16B per lane (GLOBAL_LOAD_ASYNC_TO_LDS_B128, ASYNCcnt).
// LDS byte address = low 32 bits of the generic pointer (shared aperture layout).
__device__ __forceinline__ void async_ld128(void* lds_dst, const void* gsrc) {
    unsigned loff = (unsigned)(unsigned long long)lds_dst;
    asm volatile("global_load_async_to_lds_b128 %0, %1, off"
                 :: "v"(loff), "v"(gsrc) : "memory");
}
__device__ __forceinline__ void wait_async0() {
    asm volatile("s_wait_asynccnt 0x0" ::: "memory");
}

// A fragment 16x32 f16 from LDS (row-major, 'stride' halves per row).
// Lanes 0-15: M=lane, K=0..15; lanes 16-31: K=16..31.
// Each K-octet is contiguous -> two ds_load_b128 per lane.
__device__ __forceinline__ v16h load_a_frag_lds(const _Float16* base, int stride,
                                                int mrow, int kbase) {
    int lane = threadIdx.x & 31;
    int half = lane >> 4;
    int m    = lane & 15;
    union { v16h h; unsigned int u[8]; } f;
    const _Float16* row = base + (size_t)(mrow + m) * stride + kbase;
    #pragma unroll
    for (int v = 0; v < 8; ++v) {
        int k = half * 16 + 2 * v;
        f.u[v] = *(const unsigned int*)(row + k);
    }
    return f.h;
}

// B fragment 32x16 f16 where B(k,n) = mat[(n0+n)*rowStride + koff + k]:
// per lane 16 contiguous f16 -> 32B global load.
__device__ __forceinline__ v16h load_b_frag_rows(const _Float16* mat, size_t rowStride,
                                                 int n0, int koff) {
    int lane = threadIdx.x & 31;
    int n  = lane & 15;
    int kh = (lane >> 4) * 16;
    return *(const v16h*)(mat + (size_t)(n0 + n) * rowStride + koff + kh);
}

// B fragment from LDS tile stored TRANSPOSED as [n][32 k]:
// per lane 16 contiguous f16 -> two ds_load_b128, no repacking.
__device__ __forceinline__ v16h load_b_frag_ldsT(const _Float16* ldsT, int n0) {
    int lane = threadIdx.x & 31;
    int n  = (lane & 15) + n0;
    int kh = (lane >> 4) * 16;
    return *(const v16h*)(ldsT + (size_t)n * 32 + kh);
}

__device__ __forceinline__ void store_pair(_Float16* dst, _Float16 h0, _Float16 h1) {
    union { _Float16 h[2]; unsigned u; } p;
    p.h[0] = h0; p.h[1] = h1;
    *(unsigned*)dst = p.u;
}

// ---------------------------------------------------------------------------
// 0) pad + convert tokens (B,L,H) f32 -> xh (B,S,H) f16 (zero-padded rows)
// ---------------------------------------------------------------------------
__global__ void k_pad_cvt(const float* __restrict__ tokens, _Float16* __restrict__ xh) {
    size_t i = ((size_t)blockIdx.x * 256 + threadIdx.x) * 4;
    size_t sh = (size_t)S_ * H_;
    int b = (int)(i / sh);
    size_t r = i - (size_t)b * sh;
    int s = (int)(r / H_);
    int h = (int)(r - (size_t)s * H_);
    if (s < L_) {
        const float* src = tokens + ((size_t)b * L_ + s) * H_ + h;
        #pragma unroll
        for (int q = 0; q < 4; ++q) xh[i + q] = (_Float16)src[q];
    } else {
        #pragma unroll
        for (int q = 0; q < 4; ++q) xh[i + q] = (_Float16)0.f;
    }
}

// ---------------------------------------------------------------------------
// 1) Generic WMMA GEMM: C[b] = A[b](MxK f16, strides rsA/csA) @ W(KxN f32) + bias
//    block tile 128x64, KC=32, 8 waves (4M x 2N), 2x2 wmma tiles per wave.
// ---------------------------------------------------------------------------
__global__ __launch_bounds__(256) void k_gemm_f16(
    const _Float16* __restrict__ A, size_t aBatch, int rsA, int csA,
    const float* __restrict__ W, const float* __restrict__ bias,
    _Float16* __restrict__ C, size_t cBatch,
    int M, int K, int N)
{
    int b = blockIdx.z;
    const _Float16* Ab = A + (size_t)b * aBatch;
    _Float16* Cb = C + (size_t)b * cBatch;
    int m0 = blockIdx.x * 128;
    int n0 = blockIdx.y * 64;
    int tid = threadIdx.x;
    int wid = tid >> 5, lane = tid & 31;
    int waveM = wid & 3, waveN = wid >> 2;

    __shared__ _Float16 As[128 * 32];   // [m][k]  8KB
    __shared__ _Float16 WsT[64 * 32];   // [n][k]  4KB (transposed for b-frags)

    v8f acc[2][2];
    #pragma unroll
    for (int i = 0; i < 2; ++i)
        #pragma unroll
        for (int j = 0; j < 2; ++j) acc[i][j] = vzero8();

    for (int kt = 0; kt < K; kt += 32) {
        // ---- stage A 128x32 ----
        if (csA == 1) {
            int row = tid >> 1;             // 0..127
            int c0  = (tid & 1) * 16;       // 0 or 16
            if (m0 + row < M) {
                const uvec4_t* ap = (const uvec4_t*)(Ab + (size_t)(m0 + row) * rsA + kt + c0);
                *(uvec4_t*)&As[row * 32 + c0]     = ap[0];
                *(uvec4_t*)&As[row * 32 + c0 + 8] = ap[1];
            } else {
                uvec4_t z = {0u,0u,0u,0u};
                *(uvec4_t*)&As[row * 32 + c0]     = z;
                *(uvec4_t*)&As[row * 32 + c0 + 8] = z;
            }
        } else {
            // A^T contiguous along m (rsA==1): read m-runs, scatter packed k-pairs
            int k2 = (tid >> 4) * 2;        // 0..30
            int mi = (tid & 15) * 8;        // 0..120
            uvec4_t r0 = *(const uvec4_t*)(Ab + (size_t)(kt + k2)     * csA + (size_t)(m0 + mi) * rsA);
            uvec4_t r1 = *(const uvec4_t*)(Ab + (size_t)(kt + k2 + 1) * csA + (size_t)(m0 + mi) * rsA);
            const _Float16* h0 = (const _Float16*)&r0;
            const _Float16* h1 = (const _Float16*)&r1;
            #pragma unroll
            for (int q = 0; q < 8; ++q)
                store_pair(&As[(mi + q) * 32 + k2], h0[q], h1[q]);
        }
        // ---- stage W 32x64 -> WsT[n][k] (f32->f16, packed k-pairs) ----
        {
            int k2 = (tid >> 4) * 2;        // 0..30
            int nq = (tid & 15) * 4;        // 0..60
            const float* w0 = W + (size_t)(kt + k2) * N + n0 + nq;
            const float* w1 = w0 + N;
            __builtin_prefetch(w0 + (size_t)32 * N, 0, 1);
            #pragma unroll
            for (int q = 0; q < 4; ++q)
                store_pair(&WsT[(nq + q) * 32 + k2], (_Float16)w0[q], (_Float16)w1[q]);
        }
        __syncthreads();

        v16h bf0 = load_b_frag_ldsT(WsT, waveN * 32 + 0);
        v16h bf1 = load_b_frag_ldsT(WsT, waveN * 32 + 16);
        #pragma unroll
        for (int i = 0; i < 2; ++i) {
            v16h af = load_a_frag_lds(As, 32, waveM * 32 + i * 16, 0);
            acc[i][0] = wmma_f16(af, bf0, acc[i][0]);
            acc[i][1] = wmma_f16(af, bf1, acc[i][1]);
        }
        __syncthreads();
    }

    // store C (+bias); D layout: VGPR r -> M = r + 8*(lane>=16), N = lane&15
    int n_l = lane & 15, mh = (lane >> 4) * 8;
    #pragma unroll
    for (int i = 0; i < 2; ++i) {
        #pragma unroll
        for (int j = 0; j < 2; ++j) {
            int gn = n0 + waveN * 32 + j * 16 + n_l;
            float bv = bias[gn];
            #pragma unroll
            for (int r = 0; r < 8; ++r) {
                int gm = m0 + waveM * 32 + i * 16 + r + mh;
                if (gm < M) Cb[(size_t)gm * N + gn] = (_Float16)(acc[i][j][r] + bv);
            }
        }
    }
}

// ---------------------------------------------------------------------------
// 2) Token attention: per (16-query-row tile, batch).
// ---------------------------------------------------------------------------
__global__ __launch_bounds__(256) void k_attn_tok(
    const _Float16* __restrict__ Qf, const _Float16* __restrict__ Kf,
    const _Float16* __restrict__ Vf, _Float16* __restrict__ cat)
{
    __shared__ char smem[60 * 1024];
    _Float16* sQ   = (_Float16*)smem;              // 16x768 f16 = 24KB  [0,24K)
    float*    sS   = (float*)(smem + 24 * 1024);   // 16x320 f32 = 20KB  [24K,44K)
    _Float16* sAtt = (_Float16*)(smem + 48 * 1024);// 16x320 f16 = 10KB  [48K,58K)
    float*    sRed = (float*)(smem + 58 * 1024);
    float*    sMax = sRed + 256;
    float*    sSum = sMax + 16;
    _Float16* sVT  = (_Float16*)smem;              // 768x32 f16 = 48KB (phase 2, [n][k])

    int b = blockIdx.y, q0 = blockIdx.x * 16;
    int tid = threadIdx.x, lane = tid & 31, wid = tid >> 5;
    const _Float16* Qb = Qf + ((size_t)b * S_ + q0) * H_;
    const _Float16* Kb = Kf + (size_t)b * S_ * H_;
    const _Float16* Vb = Vf + (size_t)b * S_ * H_;

    // stage Q tile asynchronously (rows contiguous): global -> LDS, no VGPR round-trip
    for (int i = tid; i < (16 * H_) / 8; i += 256)
        async_ld128(sQ + (size_t)i * 8, Qb + (size_t)i * 8);
    wait_async0();
    __syncthreads();

    // phase 1: scores (16 x 320)
    const float scale = 0.03608439182435161f;  // 1/sqrt(768)
    for (int t = wid; t < 20; t += 8) {
        v8f acc = vzero8();
        int n0 = t * 16;
        for (int kt = 0; kt < H_; kt += 32) {
            v16h af = load_a_frag_lds(sQ, H_, 0, kt);
            v16h bf = load_b_frag_rows(Kb, H_, n0, kt);
            acc = wmma_f16(af, bf, acc);
        }
        int mh = (lane >> 4) * 8, n = n0 + (lane & 15);
        #pragma unroll
        for (int r = 0; r < 8; ++r) sS[(r + mh) * S_ + n] = acc[r] * scale;
    }
    __syncthreads();

    // softmax: 16 rows x 16 groups x 20 cols
    {
        int row = tid & 15, g = tid >> 4, c0 = g * 20;
        float mx = -1e30f;
        for (int i = 0; i < 20; ++i) mx = fmaxf(mx, sS[row * S_ + c0 + i]);
        sRed[row * 16 + g] = mx;
        __syncthreads();
        if (tid < 16) {
            float m2 = -1e30f;
            for (int g2 = 0; g2 < 16; ++g2) m2 = fmaxf(m2, sRed[tid * 16 + g2]);
            sMax[tid] = m2;
        }
        __syncthreads();
        float rmax = sMax[row];
        float sm = 0.f;
        for (int i = 0; i < 20; ++i) sm += __expf(sS[row * S_ + c0 + i] - rmax);
        sRed[row * 16 + g] = sm;
        __syncthreads();
        if (tid < 16) {
            float s2 = 0.f;
            for (int g2 = 0; g2 < 16; ++g2) s2 += sRed[tid * 16 + g2];
            sSum[tid] = s2;
        }
        __syncthreads();
        float inv = 1.f / sSum[row];
        for (int i = 0; i < 20; ++i)
            sAtt[row * S_ + c0 + i] = (_Float16)(__expf(sS[row * S_ + c0 + i] - rmax) * inv);
    }
    __syncthreads();

    // phase 2: TSA(16x768) = att(16x320) @ V(320x768); V K-slabs staged transposed
    v8f acc2[6];
    #pragma unroll
    for (int j = 0; j < 6; ++j) acc2[j] = vzero8();

    for (int kt = 0; kt < S_; kt += 32) {
        __syncthreads();
        {   // stage sVT[n][k] for 32 K rows: thread = (kpair, 48-wide n group)
            int k2  = (tid & 15) * 2;
            int n0t = (tid >> 4) * 48;
            const _Float16* v0 = Vb + (size_t)(kt + k2) * H_ + n0t;
            const _Float16* v1 = v0 + H_;
            #pragma unroll
            for (int g = 0; g < 6; ++g) {
                uvec4_t r0 = *(const uvec4_t*)(v0 + g * 8);
                uvec4_t r1 = *(const uvec4_t*)(v1 + g * 8);
                const _Float16* h0 = (const _Float16*)&r0;
                const _Float16* h1 = (const _Float16*)&r1;
                #pragma unroll
                for (int q = 0; q < 8; ++q)
                    store_pair(&sVT[(size_t)(n0t + g * 8 + q) * 32 + k2], h0[q], h1[q]);
            }
        }
        __syncthreads();
        v16h af = load_a_frag_lds(sAtt, S_, 0, kt);
        #pragma unroll
        for (int j = 0; j < 6; ++j) {
            v16h bf = load_b_frag_ldsT(sVT, wid * 96 + j * 16);
            acc2[j] = wmma_f16(af, bf, acc2[j]);
        }
    }

    int n_l = lane & 15, mh = (lane >> 4) * 8;
    #pragma unroll
    for (int j = 0; j < 6; ++j) {
        int gn = wid * 96 + j * 16 + n_l;
        #pragma unroll
        for (int r = 0; r < 8; ++r) {
            int gm = q0 + r + mh;
            cat[((size_t)b * S_ + gm) * H2_ + gn] = (_Float16)acc2[j][r];
        }
    }
}

// ---------------------------------------------------------------------------
// 3) Frequency attention: per (16-H-row tile, batch), 768-wide softmax.
//    FSA^T written into cat[:, :, H:2H].
// ---------------------------------------------------------------------------
__global__ __launch_bounds__(256) void k_attn_freq(
    const _Float16* __restrict__ QN, const _Float16* __restrict__ KN,
    const _Float16* __restrict__ VN, _Float16* __restrict__ cat)
{
    __shared__ char smem[60 * 1024];
    _Float16* sQ   = (_Float16*)smem;              // 16x320 f16 = 10KB  [0,10K)
    float*    sS   = (float*)(smem + 10 * 1024);   // 16x768 f32 = 48KB  [10K,58K)
    _Float16* sAtt = (_Float16*)smem;              // 16x768 f16 = 24KB  [0,24K) after scores read
    _Float16* sVT  = (_Float16*)(smem + 24 * 1024);// 320x32 f16 = 20KB  [24K,44K) phase 2
    float*    sRed = (float*)(smem + 58 * 1024);
    float*    sMax = sRed + 256;
    float*    sSum = sMax + 16;

    int b = blockIdx.y, m0 = blockIdx.x * 16;
    int tid = threadIdx.x, lane = tid & 31, wid = tid >> 5;
    const _Float16* Qb = QN + ((size_t)b * H_ + m0) * S_;
    const _Float16* Kb = KN + (size_t)b * H_ * S_;
    const _Float16* Vb = VN + (size_t)b * H_ * S_;

    // stage QN tile asynchronously (16 x 320, rows contiguous)
    for (int i = tid; i < (16 * S_) / 8; i += 256)
        async_ld128(sQ + (size_t)i * 8, Qb + (size_t)i * 8);
    wait_async0();
    __syncthreads();

    // phase 1: scores (16 x 768)
    const float scale = 0.05590169943749474f;  // 1/sqrt(320)
    for (int t = wid; t < 48; t += 8) {
        v8f acc = vzero8();
        int n0 = t * 16;
        for (int kt = 0; kt < S_; kt += 32) {
            v16h af = load_a_frag_lds(sQ, S_, 0, kt);
            v16h bf = load_b_frag_rows(Kb, S_, n0, kt);
            acc = wmma_f16(af, bf, acc);
        }
        int mh = (lane >> 4) * 8, n = n0 + (lane & 15);
        #pragma unroll
        for (int r = 0; r < 8; ++r) sS[(r + mh) * H_ + n] = acc[r] * scale;
    }
    __syncthreads();

    // softmax over 768 cols: register-staged exp since att aliases dead LDS
    {
        int row = tid & 15, g = tid >> 4, c0 = g * 48;
        float mx = -1e30f;
        for (int i = 0; i < 48; ++i) mx = fmaxf(mx, sS[row * H_ + c0 + i]);
        sRed[row * 16 + g] = mx;
        __syncthreads();
        if (tid < 16) {
            float m2 = -1e30f;
            for (int g2 = 0; g2 < 16; ++g2) m2 = fmaxf(m2, sRed[tid * 16 + g2]);
            sMax[tid] = m2;
        }
        __syncthreads();
        float rmax = sMax[row];
        float sm = 0.f;
        for (int i = 0; i < 48; ++i) sm += __expf(sS[row * H_ + c0 + i] - rmax);
        sRed[row * 16 + g] = sm;
        __syncthreads();
        if (tid < 16) {
            float s2 = 0.f;
            for (int g2 = 0; g2 < 16; ++g2) s2 += sRed[tid * 16 + g2];
            sSum[tid] = s2;
        }
        __syncthreads();
        float inv = 1.f / sSum[row];
        float vals[48];
        for (int i = 0; i < 48; ++i)
            vals[i] = __expf(sS[row * H_ + c0 + i] - rmax) * inv;
        __syncthreads();                       // everyone done reading sS
        for (int i = 0; i < 48; ++i)
            sAtt[row * H_ + c0 + i] = (_Float16)vals[i];
    }
    __syncthreads();

    // phase 2: FSArow(16 x 320) = att(16x768) @ VN(768x320)
    int tcount = (wid < 4) ? 3 : 2;            // 20 col tiles over 8 waves
    v8f acc3[3];
    #pragma unroll
    for (int q = 0; q < 3; ++q) acc3[q] = vzero8();

    for (int kt = 0; kt < H_; kt += 32) {
        __syncthreads();
        {   // stage sVT[n][k]: thread = (kpair, 20-wide n group)
            int k2  = (tid & 15) * 2;
            int n0t = (tid >> 4) * 20;
            const _Float16* v0 = Vb + (size_t)(kt + k2) * S_ + n0t;
            const _Float16* v1 = v0 + S_;
            #pragma unroll
            for (int g = 0; g < 5; ++g) {
                uvec2_t r0 = *(const uvec2_t*)(v0 + g * 4);
                uvec2_t r1 = *(const uvec2_t*)(v1 + g * 4);
                const _Float16* h0 = (const _Float16*)&r0;
                const _Float16* h1 = (const _Float16*)&r1;
                #pragma unroll
                for (int q = 0; q < 4; ++q)
                    store_pair(&sVT[(size_t)(n0t + g * 4 + q) * 32 + k2], h0[q], h1[q]);
            }
        }
        __syncthreads();
        v16h af = load_a_frag_lds(sAtt, H_, 0, kt);
        for (int q = 0; q < tcount; ++q) {
            v16h bf = load_b_frag_ldsT(sVT, (wid + q * 8) * 16);
            acc3[q] = wmma_f16(af, bf, acc3[q]);
        }
    }

    // store transposed: result(h_row, s_col) -> cat[b, s, H + h]
    int n_l = lane & 15, mh = (lane >> 4) * 8;
    for (int q = 0; q < tcount; ++q) {
        int gs0 = (wid + q * 8) * 16 + n_l;
        #pragma unroll
        for (int r = 0; r < 8; ++r) {
            int gh = m0 + r + mh;
            cat[((size_t)b * S_ + gs0) * H2_ + H_ + gh] = (_Float16)acc3[q][r];
        }
    }
}

// ---------------------------------------------------------------------------
// 4) Pooling + assemble out_all (f16, per batch)
// ---------------------------------------------------------------------------
__global__ __launch_bounds__(256) void k_pool(
    const _Float16* __restrict__ cat, const float* __restrict__ tokens,
    const float* __restrict__ cls_out, const float* __restrict__ qry_out,
    const float* __restrict__ Wp, const float* __restrict__ bp,
    _Float16* __restrict__ out_all)
{
    int b = blockIdx.x, tid = threadIdx.x, lane = tid & 31, wid = tid >> 5;
    __shared__ float sLog[S_];
    __shared__ float sRed[256];
    const _Float16* catb = cat + (size_t)b * S_ * H2_;

    for (int s = wid; s < S_; s += 8) {
        const _Float16* row = catb + (size_t)s * H2_;
        float acc = 0.f;
        for (int c = lane; c < H2_; c += 32) acc += (float)row[c] * Wp[c];
        for (int off = 16; off; off >>= 1) acc += __shfl_down(acc, off, 32);
        if (lane == 0) sLog[s] = acc + bp[0];
    }
    __syncthreads();

    float mx = -1e30f;
    for (int s = tid; s < S_; s += 256) mx = fmaxf(mx, sLog[s]);
    sRed[tid] = mx; __syncthreads();
    for (int off = 128; off; off >>= 1) {
        if (tid < off) sRed[tid] = fmaxf(sRed[tid], sRed[tid + off]);
        __syncthreads();
    }
    mx = sRed[0]; __syncthreads();
    float sm = 0.f;
    for (int s = tid; s < S_; s += 256) sm += __expf(sLog[s] - mx);
    sRed[tid] = sm; __syncthreads();
    for (int off = 128; off; off >>= 1) {
        if (tid < off) sRed[tid] += sRed[tid + off];
        __syncthreads();
    }
    float inv = 1.f / sRed[0]; __syncthreads();
    for (int s = tid; s < S_; s += 256) sLog[s] = __expf(sLog[s] - mx) * inv;
    __syncthreads();

    _Float16* oa = out_all + (size_t)b * OA_;
    for (int c = tid; c < H2_; c += 256) {
        float a2 = 0.f;
        for (int s = 0; s < S_; ++s) a2 += sLog[s] * (float)catb[(size_t)s * H2_ + c];
        oa[H_ + c] = (_Float16)a2;
    }
    for (int h = tid; h < H_; h += 256) {
        oa[h]                 = (_Float16)tokens[(size_t)b * L_ * H_ + h];  // CLS = tokens[b,0,:]
        oa[H_ + H2_ + h]      = (_Float16)cls_out[(size_t)b * H_ + h];
        oa[H_ + H2_ + H_ + h] = (_Float16)qry_out[(size_t)b * H_ + h];
    }
}

// ---------------------------------------------------------------------------
// 5) final tiny GEMM: out(64,3) = hidden(64,768) @ W2(768,3) + b2
// ---------------------------------------------------------------------------
__global__ void k_final(const _Float16* __restrict__ hidden, const float* __restrict__ W2,
                        const float* __restrict__ b2, float* __restrict__ out) {
    int i = blockIdx.x * blockDim.x + threadIdx.x;
    if (i >= B_ * 3) return;
    int b = i / 3, c = i % 3;
    float acc = b2[c];
    const _Float16* h = hidden + (size_t)b * H_;
    for (int k = 0; k < H_; ++k) acc += (float)h[k] * W2[k * 3 + c];
    out[i] = acc;
}

// ---------------------------------------------------------------------------
extern "C" void kernel_launch(void* const* d_in, const int* in_sizes, int n_in,
                              void* d_out, int out_size, void* d_ws, size_t ws_size,
                              hipStream_t stream) {
    (void)in_sizes; (void)n_in; (void)out_size; (void)ws_size;
    const float* tokens   = (const float*)d_in[0];
    const float* cls_out  = (const float*)d_in[1];
    const float* qry_out  = (const float*)d_in[2];
    const float* Wq  = (const float*)d_in[3];   const float* bq  = (const float*)d_in[4];
    const float* Wk  = (const float*)d_in[5];   const float* bk  = (const float*)d_in[6];
    const float* Wv  = (const float*)d_in[7];   const float* bv  = (const float*)d_in[8];
    const float* Wfq = (const float*)d_in[9];   const float* bfq = (const float*)d_in[10];
    const float* Wfk = (const float*)d_in[11];  const float* bfk = (const float*)d_in[12];
    const float* Wfv = (const float*)d_in[13];  const float* bfv = (const float*)d_in[14];
    const float* Wp  = (const float*)d_in[15];  const float* bp  = (const float*)d_in[16];
    const float* W1  = (const float*)d_in[17];  const float* b1  = (const float*)d_in[18];
    const float* W2  = (const float*)d_in[19];  const float* b2  = (const float*)d_in[20];
    float* out = (float*)d_out;

    const size_t SZ_BSH = (size_t)B_ * S_ * H_;
    _Float16* xh  = (_Float16*)d_ws;
    _Float16* Qf  = xh  + SZ_BSH;
    _Float16* Kf  = Qf  + SZ_BSH;
    _Float16* Vf  = Kf  + SZ_BSH;
    _Float16* QNf = Vf  + SZ_BSH;
    _Float16* KNf = QNf + SZ_BSH;
    _Float16* VNf = KNf + SZ_BSH;
    _Float16* cat = VNf + SZ_BSH;                       // B*S*2H
    _Float16* oa  = cat + (size_t)B_ * S_ * H2_;        // B*5H
    _Float16* hid = oa  + (size_t)B_ * OA_;             // B*H

    dim3 blk(256);

    k_pad_cvt<<<dim3((unsigned)(SZ_BSH / (256 * 4))), blk, 0, stream>>>(tokens, xh);

    // QKV projections: (B) x [320x768x768], contiguous-A path
    k_gemm_f16<<<dim3(3, 12, B_), blk, 0, stream>>>(xh, SZ_BSH, H_, 1, Wq, bq, Qf, SZ_BSH, S_, H_, H_);
    k_gemm_f16<<<dim3(3, 12, B_), blk, 0, stream>>>(xh, SZ_BSH, H_, 1, Wk, bk, Kf, SZ_BSH, S_, H_, H_);
    k_gemm_f16<<<dim3(3, 12, B_), blk, 0, stream>>>(xh, SZ_BSH, H_, 1, Wv, bv, Vf, SZ_BSH, S_, H_, H_);

    // freq projections (A = x^T via strides, rsA==1 path): (B) x [768x320x320]
    k_gemm_f16<<<dim3(6, 5, B_), blk, 0, stream>>>(xh, SZ_BSH, 1, H_, Wfq, bfq, QNf, SZ_BSH, H_, S_, S_);
    k_gemm_f16<<<dim3(6, 5, B_), blk, 0, stream>>>(xh, SZ_BSH, 1, H_, Wfk, bfk, KNf, SZ_BSH, H_, S_, S_);
    k_gemm_f16<<<dim3(6, 5, B_), blk, 0, stream>>>(xh, SZ_BSH, 1, H_, Wfv, bfv, VNf, SZ_BSH, H_, S_, S_);

    // attentions -> cat
    k_attn_tok <<<dim3(S_ / 16, B_), blk, 0, stream>>>(Qf, Kf, Vf, cat);
    k_attn_freq<<<dim3(H_ / 16, B_), blk, 0, stream>>>(QNf, KNf, VNf, cat);

    // pooling + out_all
    k_pool<<<dim3(B_), blk, 0, stream>>>(cat, tokens, cls_out, qry_out, Wp, bp, oa);

    // hidden = out_all @ W1 + b1  (64 x 3840 x 768)
    k_gemm_f16<<<dim3(1, 12, 1), blk, 0, stream>>>(oa, 0, OA_, 1, W1, b1, hid, 0, B_, OA_, H_);

    k_final<<<dim3(1), dim3(192), 0, stream>>>(hid, W2, b2, out);
}
